// VectorQuantiser_30794915512876
// MI455X (gfx1250) — matrix operand order repaired and checked by reading, not compiled, not run on hardware
//
#include <hip/hip_runtime.h>

// ---------------------------------------------------------------------------
// VQ-VAE vector quantiser for MI455X (gfx1250, wave32, WMMA).
// B=8, C=64, H=64, W=64, NUM_EMBED=8192, BETA=0.25
// N = 32768 tokens, d = 64, K = 8192 codes.
// Core: fused bf16 WMMA GEMM (16x16x32) + packed-key in-register argmax,
//       double-buffered B-fragment loads + L0 prefetch.
// ---------------------------------------------------------------------------

#define NTOK   32768      // B*H*W
#define NCODE  8192
#define DIM    64
#define HW     4096       // H*W
#define NELEM  2097152    // B*C*H*W

typedef __attribute__((ext_vector_type(16))) __bf16 v16bf;
typedef __attribute__((ext_vector_type(8)))  float  v8f;

union FragAB { v16bf v; uint4 u[2]; };

__device__ __forceinline__ unsigned short f2bf(float f) {
    // round-to-nearest-even f32 -> bf16
    unsigned u = __float_as_uint(f);
    unsigned r = u + 0x7FFFu + ((u >> 16) & 1u);
    return (unsigned short)(r >> 16);
}

// --------------------------- init accumulators -----------------------------
__global__ void vq_zero(unsigned* counts, float* lossAcc) {
    int i = blockIdx.x * blockDim.x + threadIdx.x;
    if (i < NCODE) counts[i] = 0u;
    if (i == NCODE) *lossAcc = 0.0f;
}

// ------------------- normalize codebook rows -> bf16 -----------------------
// One wave per row; lane handles 2 consecutive channels.
__global__ __launch_bounds__(256) void vq_norm_embed(const float* __restrict__ emb,
                                                     unsigned short* __restrict__ en_bf) {
    int lane = threadIdx.x & 31;
    int wave = threadIdx.x >> 5;
    int row  = blockIdx.x * 8 + wave;            // grid = NCODE/8 blocks
    const float* e = emb + (size_t)row * DIM;
    float x0 = e[2 * lane + 0];
    float x1 = e[2 * lane + 1];
    float s = x0 * x0 + x1 * x1;
    #pragma unroll
    for (int off = 16; off >= 1; off >>= 1) s += __shfl_xor(s, off, 32);
    float rn = rsqrtf(s + 1e-30f);
    unsigned pack = (unsigned)f2bf(x0 * rn) | ((unsigned)f2bf(x1 * rn) << 16);
    ((unsigned*)en_bf)[row * (DIM / 2) + lane] = pack;
}

// ------------------- normalize z token rows -> bf16 ------------------------
// Token n = (b*H + h)*W + w; channel c lives at z[(b*64 + c)*4096 + hw].
__global__ __launch_bounds__(256) void vq_norm_z(const float* __restrict__ z,
                                                 unsigned short* __restrict__ zn_bf) {
    int lane  = threadIdx.x & 31;
    int wave  = threadIdx.x >> 5;
    int token = blockIdx.x * 8 + wave;           // grid = NTOK/8 blocks
    int b  = token >> 12;
    int hw = token & (HW - 1);
    const float* zb = z + (size_t)b * DIM * HW + hw;
    float x0 = zb[(size_t)(2 * lane + 0) * HW];
    float x1 = zb[(size_t)(2 * lane + 1) * HW];
    float s = x0 * x0 + x1 * x1;
    #pragma unroll
    for (int off = 16; off >= 1; off >>= 1) s += __shfl_xor(s, off, 32);
    float rn = rsqrtf(s + 1e-30f);
    unsigned pack = (unsigned)f2bf(x0 * rn) | ((unsigned)f2bf(x1 * rn) << 16);
    ((unsigned*)zn_bf)[token * (DIM / 2) + lane] = pack;
}

// --------------- fused bf16 WMMA GEMM + per-row argmax ---------------------
// Wave handles 16 tokens. Loop over 512 chunks of 16 codes:
//   scores(16x16) = A(16x64) * B(64x16) via two v_wmma_f32_16x16x32_bf16.
// C layout: reg r, lane l -> row r + 8*(l>=16), col l%16.
// Argmax key: bits(score+2) is monotone (score in [-1,1] -> +2 positive);
// low 13 mantissa bits replaced by (8191 - col) so umax picks the best score
// and, on quantized ties, the smallest column index (jnp.argmax semantics).
__global__ __launch_bounds__(256) void vq_argmax(const unsigned short* __restrict__ zn_bf,
                                                 const unsigned short* __restrict__ en_bf,
                                                 int* __restrict__ idxOut) {
    int lane = threadIdx.x & 31;
    int wave = threadIdx.x >> 5;
    int tile = blockIdx.x * 8 + wave;            // grid = (NTOK/16)/8 = 256 blocks
    int tokenBase = tile * 16;
    int half = lane >> 4;                        // 0: lanes 0-15, 1: lanes 16-31
    int l15  = lane & 15;

    // A fragments (ISA 16-bit A 16x32 layout), kept in registers for all chunks.
    const uint4* zrow = (const uint4*)(zn_bf + (size_t)(tokenBase + l15) * DIM);
    FragAB a0, a1;
    a0.u[0] = zrow[0 + half];    // K =  8*half .. +7
    a0.u[1] = zrow[2 + half];    // K = 16 + 8*half .. +7
    a1.u[0] = zrow[4 + half];    // K = 32 + 8*half .. +7
    a1.u[1] = zrow[6 + half];    // K = 48 + 8*half .. +7

    auto loadB = [&](FragAB& f0, FragAB& f1, int c) {
        // B fragment (B 32x16 layout): col = l15, K order matches A (8*half, 16+8*half)
        const uint4* erow = (const uint4*)(en_bf + (size_t)(c * 16 + l15) * DIM);
        f0.u[0] = erow[0 + half];         // K =  8*half .. +7
        f0.u[1] = erow[2 + half];         // K = 16 + 8*half .. +7
        f1.u[0] = erow[4 + half];         // K = 32 + 8*half .. +7
        f1.u[1] = erow[6 + half];         // K = 48 + 8*half .. +7
    };

    unsigned key[8];
    #pragma unroll
    for (int r = 0; r < 8; ++r) key[r] = 0u;

    unsigned inv = 8191u - (unsigned)l15;        // 8191 - colIdx for this lane

    // Double-buffered B fragments: next chunk's loads in flight during WMMA.
    FragAB b0[2], b1[2];
    loadB(b0[0], b1[0], 0);

    #pragma unroll 2
    for (int c = 0; c < NCODE / 16; ++c) {
        int cur = c & 1;
        int nxt = cur ^ 1;
        int cn  = (c + 1 < NCODE / 16) ? (c + 1) : c;   // uniform clamp
        loadB(b0[nxt], b1[nxt], cn);

        // L0 prefetch a few chunks ahead (global_prefetch_b8).
        __builtin_prefetch(en_bf + (size_t)(((c + 6) & 511) * 16 + l15) * DIM, 0, 3);

        v8f acc = {};
        acc = __builtin_amdgcn_wmma_f32_16x16x32_bf16(false, a0.v, false, b0[cur].v,
                                                      (short)0, acc, false, false);
        acc = __builtin_amdgcn_wmma_f32_16x16x32_bf16(false, a1.v, false, b1[cur].v,
                                                      (short)0, acc, false, false);

        #pragma unroll
        for (int r = 0; r < 8; ++r) {
            unsigned kb = (__float_as_uint(acc[r] + 2.0f) & 0xFFFFE000u) | inv;
            key[r] = key[r] >= kb ? key[r] : kb;        // v_max_u32
        }
        inv -= 16u;
    }

    // Branchless butterfly max across the 16 lanes of each half (row's cols).
    #pragma unroll
    for (int off = 8; off >= 1; off >>= 1) {
        #pragma unroll
        for (int r = 0; r < 8; ++r) {
            unsigned o = (unsigned)__shfl_xor((int)key[r], off, 16);
            key[r] = key[r] >= o ? key[r] : o;
        }
    }

    if (l15 == 0) {
        int rowBase = tokenBase + half * 8;      // lanes 0 & 16 write 8 rows each
        #pragma unroll
        for (int r = 0; r < 8; ++r)
            idxOut[rowBase + r] = (int)(8191u - (key[r] & 8191u));
    }
}

// ------------- gather z_q, MSE loss accumulation, histogram ----------------
__global__ __launch_bounds__(256) void vq_gather(const float* __restrict__ z,
                                                 const float* __restrict__ emb,
                                                 const int* __restrict__ idx,
                                                 float* __restrict__ out,
                                                 float* __restrict__ idxMap,
                                                 unsigned* __restrict__ counts,
                                                 float* __restrict__ lossAcc) {
    int n = blockIdx.x * blockDim.x + threadIdx.x;   // grid = NTOK/256
    int k = idx[n];
    int b  = n >> 12;
    int hw = n & (HW - 1);
    const float* er = emb + (size_t)k * DIM;
    float s = 0.0f;
    #pragma unroll 8
    for (int c = 0; c < DIM; ++c) {
        float q = er[c];
        size_t a = ((size_t)(b * DIM + c)) * HW + hw; // coalesced across threads
        float d = q - z[a];
        s += d * d;
        out[a] = q;                                   // straight-through forward == z_q
    }
    atomicAdd(lossAcc, s);
    atomicAdd(&counts[k], 1u);
    idxMap[n] = (float)k;
}

// ------------------------ loss + perplexity finalize -----------------------
__global__ __launch_bounds__(256) void vq_final(const unsigned* __restrict__ counts,
                                                const float* __restrict__ lossAcc,
                                                float* __restrict__ outLoss,
                                                float* __restrict__ outPerp) {
    __shared__ float sh[256];
    float s = 0.0f;
    for (int i = threadIdx.x; i < NCODE; i += 256) {
        float p = (float)counts[i] * (1.0f / (float)NTOK);
        s += p * logf(p + 1e-10f);
    }
    sh[threadIdx.x] = s;
    __syncthreads();
    for (int st = 128; st > 0; st >>= 1) {
        if ((int)threadIdx.x < st) sh[threadIdx.x] += sh[threadIdx.x + st];
        __syncthreads();
    }
    if (threadIdx.x == 0) {
        *outPerp = expf(-sh[0]);
        // loss = beta*mean + mean = 1.25 * sum / (B*C*H*W)
        *outLoss = 1.25f * (*lossAcc) / (float)NELEM;
    }
}

// ---------------------------------------------------------------------------
extern "C" void kernel_launch(void* const* d_in, const int* in_sizes, int n_in,
                              void* d_out, int out_size, void* d_ws, size_t ws_size,
                              hipStream_t stream) {
    const float* z   = (const float*)d_in[0];   // [8,64,64,64]
    const float* emb = (const float*)d_in[1];   // [8192,64]

    // Workspace layout
    unsigned short* en_bf = (unsigned short*)d_ws;              // 8192*64  (1 MB)
    unsigned short* zn_bf = en_bf + (size_t)NCODE * DIM;        // 32768*64 (4 MB)
    int*            idx   = (int*)(zn_bf + (size_t)NTOK * DIM); // 32768*4
    unsigned*     counts  = (unsigned*)(idx + NTOK);            // 8192*4
    float*       lossAcc  = (float*)(counts + NCODE);           // 4

    // Output layout: out | loss | perplexity | idx_map
    float* out     = (float*)d_out;
    float* outLoss = out + NELEM;
    float* outPerp = out + NELEM + 1;
    float* idxMap  = out + NELEM + 2;

    vq_zero<<<(NCODE + 256) / 256, 256, 0, stream>>>(counts, lossAcc);
    vq_norm_embed<<<NCODE / 8, 256, 0, stream>>>(emb, en_bf);
    vq_norm_z<<<NTOK / 8, 256, 0, stream>>>(z, zn_bf);
    vq_argmax<<<(NTOK / 16) / 8, 256, 0, stream>>>(zn_bf, en_bf, idx);
    vq_gather<<<NTOK / 256, 256, 0, stream>>>(z, emb, idx, out, idxMap, counts, lossAcc);
    vq_final<<<1, 256, 0, stream>>>(counts, lossAcc, outLoss, outPerp);

    (void)in_sizes; (void)n_in; (void)out_size; (void)ws_size;
}
